// GaussianRefineHeadSparseConv_51316269253313
// MI455X (gfx1250) — matrix operand order repaired and checked by reading, not compiled, not run on hardware
//
#include <hip/hip_runtime.h>
#include <hip/hip_bf16.h>

typedef __attribute__((ext_vector_type(16))) __bf16 v16bf;
typedef __attribute__((ext_vector_type(8)))  __bf16 v8bf;
typedef __attribute__((ext_vector_type(8)))  float  v8f;

#define NPTS   131072
#define GDIM   14
#define FEAT   128
#define NHID   64
#define NTAP   27
#define NLAY   2
#define CAP    262144
#define CAPM   (CAP - 1)
#define VOXSZ  0.05f
#define EPSF   1e-5f

// header word indices (4KB header at ws+0; hdri = int view, hdrf = float view)
#define HI_VCMIN 0
#define HI_VCMAX 3
#define HI_MX    6
#define HI_M12   9
#define HI_MCNT  10
#define HF_PSUM  11
#define HF_PS    14
#define HF_BNSUM 64
#define HF_BNSQ  192

static __device__ __forceinline__ __bf16 f2bf(float f) {
  union { unsigned short u; __bf16 b; } cv;
  unsigned u = __float_as_uint(f);
  unsigned r = u + 0x7FFFu + ((u >> 16) & 1u);   // round-to-nearest-even
  cv.u = (unsigned short)(r >> 16);
  return cv.b;
}

static __device__ __forceinline__ int probe_table(const int* tkey, const int* tval, int h) {
  unsigned slot = ((unsigned)h * 2654435761u) & CAPM;
  for (;;) {
    int kk = tkey[slot];
    if (kk == h)  return tval[slot];
    if (kk == -1) return -1;
    slot = (slot + 1u) & CAPM;
  }
}

// ---------------- setup kernels ----------------

__global__ void k_init(const float* ps, int* hdri, float* hdrf) {
  for (int a = 0; a < 3; a++) { hdri[HI_VCMIN + a] = 0x7FFFFFFF; hdri[HI_VCMAX + a] = (int)0x80000000; }
  hdri[HI_MCNT] = 0;
  hdrf[HF_PSUM + 0] = 0.f; hdrf[HF_PSUM + 1] = 0.f; hdrf[HF_PSUM + 2] = 0.f;
  float p = ps[0];
  bool ok = (p == p) && !__builtin_isinf(p) && (p != 0.f);
  hdrf[HF_PS] = ok ? p : 1.f;
}

__global__ void k_prep(const float* __restrict__ gp, float* __restrict__ gpc, float* hdrf) {
  int p = blockIdx.x * blockDim.x + threadIdx.x;
  if (p >= NPTS) return;
  float v[GDIM];
  #pragma unroll
  for (int k = 0; k < GDIM; k++) {
    float x = gp[(size_t)p * GDIM + k];
    if (x != x) x = 0.f;
    else if (__builtin_isinf(x)) x = (x > 0.f) ? 1.f : -1.f;
    v[k] = x;
    gpc[(size_t)p * GDIM + k] = x;
  }
  float psi = 1.0f / hdrf[HF_PS];
  atomicAdd(&hdrf[HF_PSUM + 0], v[0] * psi);
  atomicAdd(&hdrf[HF_PSUM + 1], v[1] * psi);
  atomicAdd(&hdrf[HF_PSUM + 2], v[2] * psi);
}

__global__ void k_vox(const float* __restrict__ gpc, const float* hdrf, int* hdri, int* __restrict__ vc) {
  int p = blockIdx.x * blockDim.x + threadIdx.x;
  if (p >= NPTS) return;
  float psi = 1.0f / hdrf[HF_PS];
  #pragma unroll
  for (int a = 0; a < 3; a++) {
    float mean = hdrf[HF_PSUM + a] * (1.0f / NPTS);
    float pos = gpc[(size_t)p * GDIM + a] * psi - mean;
    int c = (int)floorf(pos / VOXSZ);
    vc[p * 3 + a] = c;
    atomicMin(&hdri[HI_VCMIN + a], c);
    atomicMax(&hdri[HI_VCMAX + a], c);
  }
}

__global__ void k_grid(int* hdri) {
  int mx[3];
  for (int a = 0; a < 3; a++) { mx[a] = hdri[HI_VCMAX + a] - hdri[HI_VCMIN + a] + 1; hdri[HI_MX + a] = mx[a]; }
  hdri[HI_M12] = mx[1] * mx[2];
}

__global__ void k_hash(const int* __restrict__ vc, int* hdri, int* __restrict__ hpt,
                       int* __restrict__ tkey, int* __restrict__ tval, int* __restrict__ vhash) {
  int p = blockIdx.x * blockDim.x + threadIdx.x;
  if (p >= NPTS) return;
  int c0 = vc[p * 3 + 0] - hdri[HI_VCMIN + 0];
  int c1 = vc[p * 3 + 1] - hdri[HI_VCMIN + 1];
  int c2 = vc[p * 3 + 2] - hdri[HI_VCMIN + 2];
  int h = c0 * hdri[HI_M12] + c1 * hdri[HI_MX + 2] + c2;
  hpt[p] = h;
  unsigned slot = ((unsigned)h * 2654435761u) & CAPM;
  for (;;) {
    int prev = atomicCAS(&tkey[slot], -1, h);
    if (prev == -1) {               // this thread claimed the voxel
      int vid = atomicAdd(&hdri[HI_MCNT], 1);
      vhash[vid] = h;
      tval[slot] = vid;
      break;
    }
    if (prev == h) break;           // voxel exists / being created
    slot = (slot + 1u) & CAPM;
  }
}

__global__ void k_inv(const int* __restrict__ hpt, const int* __restrict__ tkey,
                      const int* __restrict__ tval, int* __restrict__ inv) {
  int p = blockIdx.x * blockDim.x + threadIdx.x;
  if (p >= NPTS) return;
  inv[p] = probe_table(tkey, tval, hpt[p]);
}

// encoder: f = relu(LN(gp @ w_in + b_in)); scatter-mean into voxel accumulators
__global__ void k_enc(const float* __restrict__ gpc, const float* __restrict__ w_in,
                      const float* __restrict__ b_in, const float* __restrict__ ln_g,
                      const float* __restrict__ ln_b, const int* __restrict__ inv,
                      float* __restrict__ acc, float* __restrict__ vcnt) {
  __shared__ float sg[GDIM];
  __shared__ float red[FEAT];
  int p = blockIdx.x;
  int j = threadIdx.x;
  if (j < GDIM) sg[j] = gpc[(size_t)p * GDIM + j];
  __syncthreads();
  float f = b_in[j];
  #pragma unroll
  for (int k = 0; k < GDIM; k++) f += sg[k] * w_in[k * FEAT + j];
  red[j] = f;
  __syncthreads();
  for (int s = FEAT / 2; s > 0; s >>= 1) { if (j < s) red[j] += red[j + s]; __syncthreads(); }
  float mu = red[0] * (1.0f / FEAT);
  __syncthreads();
  float d = f - mu;
  red[j] = d * d;
  __syncthreads();
  for (int s = FEAT / 2; s > 0; s >>= 1) { if (j < s) red[j] += red[j + s]; __syncthreads(); }
  float var = red[0] * (1.0f / FEAT);
  float y = d * rsqrtf(var + EPSF) * ln_g[j] + ln_b[j];
  y = fmaxf(y, 0.0f);
  int m = inv[p];
  atomicAdd(&acc[(size_t)m * FEAT + j], y);
  if (j == 0) atomicAdd(&vcnt[m], 1.0f);
}

__global__ void k_vfnorm(const float* __restrict__ acc, const float* __restrict__ vcnt,
                         const int* hdri, float* __restrict__ vf32, __bf16* __restrict__ vfb) {
  int idx = blockIdx.x * blockDim.x + threadIdx.x;
  int M = hdri[HI_MCNT];
  if (idx >= M * FEAT) return;
  int m = idx >> 7;
  float v = acc[idx] / fmaxf(vcnt[m], 1.0f);
  vf32[idx] = v;
  vfb[idx] = f2bf(v);
}

__global__ void k_nbr(const int* __restrict__ vhash, const int* __restrict__ tkey,
                      const int* __restrict__ tval, const int* hdri, int* __restrict__ nbr) {
  int m = blockIdx.x * blockDim.x + threadIdx.x;
  int M = hdri[HI_MCNT];
  if (m >= M) return;
  int m12 = hdri[HI_M12], mx0 = hdri[HI_MX + 0], mx1 = hdri[HI_MX + 1], mx2 = hdri[HI_MX + 2];
  int h = vhash[m];
  int cz = h / m12, rem = h % m12;
  int cy = rem / mx2, cx = rem % mx2;
  int k = 0;
  for (int dz = -1; dz <= 1; dz++)
    for (int dy = -1; dy <= 1; dy++)
      for (int dx = -1; dx <= 1; dx++) {
        int nz = cz + dz, ny = cy + dy, nx = cx + dx;
        int res = -1;
        if (nz >= 0 && nz < mx0 && ny >= 0 && ny < mx1 && nx >= 0 && nx < mx2)
          res = probe_table(tkey, tval, nz * m12 + ny * mx2 + nx);
        nbr[(size_t)m * NTAP + (k++)] = res;
      }
}

// Pre-swizzle conv weights into WMMA B-fragment layout:
//   Wp[kt][col][kb][e] = bf16( W[kt][kb*16 + e][col] ),  kb = K/16 block, e = K within block.
// A lane's B fragment for (kstep ks, lane-half h) is then the contiguous 32-byte
// strip Wp[kt][ncol][2*ks + h][0..15]  -> one aligned v16bf load (2x global_load_b128).
__global__ void k_wpack(const float* __restrict__ w1, const float* __restrict__ w2,
                        __bf16* __restrict__ o1, __bf16* __restrict__ o2) {
  int idx = blockIdx.x * blockDim.x + threadIdx.x;
  if (idx >= NLAY * NTAP * FEAT * FEAT) return;
  int e   = idx & 15;
  int kb  = (idx >> 4) & 7;
  int col = (idx >> 7) & (FEAT - 1);
  int kt  = idx >> 14;                       // combined (layer*27 + tap)
  size_t src = ((size_t)kt * FEAT + (kb * 16 + e)) * FEAT + col;
  o1[idx] = f2bf(w1[src]);
  o2[idx] = f2bf(w2[src]);
}

// ---------------- WMMA submanifold conv ----------------
// Block = 256 threads (8 wave32s). Block owns a 16-voxel M-tile x full N=128.
// Wave w owns N-tile [16w,16w+16). Per tap: stage 16 gathered bf16 rows to LDS
// (1x uint4 per thread), then 4 k-steps of v_wmma_f32_16x16x32_bf16 fed by
// 2x ds_load_b128 (A) + 2x global_load_b128 (pre-swizzled B).
__global__ __launch_bounds__(256) void k_conv(const __bf16* __restrict__ vin,
                                              const __bf16* __restrict__ wp,
                                              const int* __restrict__ nbr,
                                              const int* hdri,
                                              float* __restrict__ out) {
  __shared__ __attribute__((aligned(16))) __bf16 sA[16][136];  // 272B row stride = 17*16B
  __shared__ int sN[16];
  int M = hdri[HI_MCNT];
  int m0 = blockIdx.x * 16;
  if (m0 >= M) return;
  int tid = threadIdx.x;
  int wave = tid >> 5, lane = tid & 31;
  int n0 = wave * 16;
  int half = lane >> 4, mrow = lane & 15, ncol = n0 + (lane & 15);
  int srow = tid >> 4, sc4 = tid & 15;       // staging: 16B chunk per thread
  v8f acc = {0.f, 0.f, 0.f, 0.f, 0.f, 0.f, 0.f, 0.f};

  for (int k = 0; k < NTAP; k++) {
    if (tid < 16) {
      int m = m0 + tid;
      sN[tid] = (m < M) ? nbr[(size_t)m * NTAP + k] : -1;
    }
    if (k + 1 < NTAP) __builtin_prefetch(wp + (size_t)(k + 1) * FEAT * FEAT, 0, 3);
    __syncthreads();
    // stage 16 rows x 256B: one uint4 load + ds_store_b128 per thread
    {
      int nb = sN[srow];
      uint4 val = make_uint4(0u, 0u, 0u, 0u);
      if (nb >= 0) val = ((const uint4*)vin)[(size_t)nb * 16 + sc4];
      *(uint4*)&sA[srow][sc4 * 8] = val;
    }
    __syncthreads();
    const v16bf* Wt = (const v16bf*)wp + ((size_t)k * FEAT + ncol) * 8;
    #pragma unroll
    for (int ks = 0; ks < 4; ks++) {
      int c0 = ks * 32 + 16 * half;
      // A 16x32 bf16 fragment (contiguous k = 16*half + s, matching B): two aligned 16B chunks per lane
      v8bf alo = *(const v8bf*)&sA[mrow][c0];
      v8bf ahi = *(const v8bf*)&sA[mrow][c0 + 8];
      v16bf a = __builtin_shufflevector(alo, ahi, 0, 1, 2, 3, 4, 5, 6, 7,
                                        8, 9, 10, 11, 12, 13, 14, 15);
      // B 32x16 bf16 fragment: one contiguous pre-swizzled 32B strip
      v16bf b = Wt[ks * 2 + half];
      acc = __builtin_amdgcn_wmma_f32_16x16x32_bf16(false, a, false, b, (short)0, acc, false, false);
    }
    __syncthreads();
  }
  // D 16x16 f32: lanes 0-15: N=lane, M=v; lanes 16-31: N=lane-16, M=v+8
  #pragma unroll
  for (int v = 0; v < 8; v++) {
    int row = m0 + v + 8 * half;
    if (row < M) out[(size_t)row * FEAT + n0 + (lane & 15)] = acc[v];
  }
}

// ---------------- batchnorm ----------------

__global__ void k_bnstats(const float* __restrict__ hb, const int* hdri, float* hdrf) {
  int M = hdri[HI_MCNT];
  int f = threadIdx.x;             // 128 threads
  int r0 = blockIdx.x * 64;
  if (r0 >= M) return;
  float s = 0.f, q = 0.f;
  for (int r = 0; r < 64; r++) {
    int m = r0 + r;
    if (m < M) { float v = hb[(size_t)m * FEAT + f]; s += v; q += v * v; }
  }
  atomicAdd(&hdrf[HF_BNSUM + f], s);
  atomicAdd(&hdrf[HF_BNSQ + f], q);
}

__global__ void k_bnapply(const float* __restrict__ hb, const float* __restrict__ g,
                          const float* __restrict__ bta, const int* hdri, const float* hdrf,
                          const float* __restrict__ resid, float* __restrict__ out32,
                          __bf16* __restrict__ outb) {
  int idx = blockIdx.x * blockDim.x + threadIdx.x;
  int M = hdri[HI_MCNT];
  if (idx >= M * FEAT) return;
  int f = idx & (FEAT - 1);
  float Mf = fmaxf((float)M, 1.0f);
  float mean = hdrf[HF_BNSUM + f] / Mf;
  float var  = hdrf[HF_BNSQ + f] / Mf - mean * mean;
  float y = (hb[idx] - mean) * rsqrtf(fmaxf(var, 0.f) + EPSF) * g[f] + bta[f];
  if (resid) y += resid[idx];
  y = fmaxf(y, 0.0f);
  if (out32) out32[idx] = y;
  outb[idx] = f2bf(y);
}

// ---------------- output head ----------------

__global__ void k_head(const float* __restrict__ vf32, const int* __restrict__ inv,
                       const float* __restrict__ w1, const float* __restrict__ b1,
                       const float* __restrict__ w2, const float* __restrict__ b2,
                       float* __restrict__ out) {
  __shared__ float pf[FEAT];
  __shared__ float hid[NHID];
  int p = blockIdx.x;
  int j = threadIdx.x;             // 64 threads
  int m = inv[p];
  pf[j]      = vf32[(size_t)m * FEAT + j];
  pf[j + 64] = vf32[(size_t)m * FEAT + j + 64];
  __syncthreads();
  float h = b1[j];
  #pragma unroll 8
  for (int c = 0; c < FEAT; c++) h += pf[c] * w1[c * NHID + j];
  hid[j] = fmaxf(h, 0.0f);
  __syncthreads();
  if (j < GDIM) {
    float o = b2[j];
    #pragma unroll 8
    for (int c = 0; c < NHID; c++) o += hid[c] * w2[c * GDIM + j];
    out[(size_t)p * GDIM + j] = o;
  }
}

// ---------------- launch ----------------

extern "C" void kernel_launch(void* const* d_in, const int* in_sizes, int n_in,
                              void* d_out, int out_size, void* d_ws, size_t ws_size,
                              hipStream_t stream) {
  (void)in_sizes; (void)n_in; (void)out_size; (void)ws_size;
  const float* gp   = (const float*)d_in[0];
  const float* ps   = (const float*)d_in[1];
  const float* w_in = (const float*)d_in[2];
  const float* b_in = (const float*)d_in[3];
  const float* ln_g = (const float*)d_in[4];
  const float* ln_b = (const float*)d_in[5];
  const float* cw1  = (const float*)d_in[6];
  const float* bn1g = (const float*)d_in[7];
  const float* bn1b = (const float*)d_in[8];
  const float* cw2  = (const float*)d_in[9];
  const float* bn2g = (const float*)d_in[10];
  const float* bn2b = (const float*)d_in[11];
  const float* wh1  = (const float*)d_in[12];
  const float* bh1  = (const float*)d_in[13];
  const float* wh2  = (const float*)d_in[14];
  const float* bh2  = (const float*)d_in[15];
  float* out = (float*)d_out;

  char* ws = (char*)d_ws;
  size_t off = 0;
  auto take = [&](size_t bytes) -> char* {
    char* p = ws + off;
    off += (bytes + 255) & ~(size_t)255;
    return p;
  };
  float*  hdrf = (float*)take(4096);
  int*    hdri = (int*)hdrf;
  float*  gpc  = (float*)take((size_t)NPTS * GDIM * 4);
  int*    vc   = (int*)take((size_t)NPTS * 3 * 4);
  int*    hpt  = (int*)take((size_t)NPTS * 4);
  int*    inv  = (int*)take((size_t)NPTS * 4);
  int*    tkey = (int*)take((size_t)CAP * 4);
  int*    tval = (int*)take((size_t)CAP * 4);   // contiguous after tkey
  int*    vhash= (int*)take((size_t)NPTS * 4);
  float*  vcnt = (float*)take((size_t)NPTS * 4);
  float*  acc  = (float*)take((size_t)NPTS * FEAT * 4);   // pooling accum, reused as conv out
  float*  vf32 = (float*)take((size_t)NPTS * FEAT * 4);
  __bf16* vfb  = (__bf16*)take((size_t)NPTS * FEAT * 2);
  __bf16* hb16 = (__bf16*)take((size_t)NPTS * FEAT * 2);
  int*    nbr  = (int*)take((size_t)NPTS * NTAP * 4);
  __bf16* wb1  = (__bf16*)take((size_t)NLAY * NTAP * FEAT * FEAT * 2);
  __bf16* wb2  = (__bf16*)take((size_t)NLAY * NTAP * FEAT * FEAT * 2);

  hipMemsetAsync(tkey, 0xFF, (size_t)CAP * 4 * 2, stream);     // keys+vals -> -1
  hipMemsetAsync(acc, 0, (size_t)NPTS * FEAT * 4, stream);
  hipMemsetAsync(vcnt, 0, (size_t)NPTS * 4, stream);

  k_init<<<1, 1, 0, stream>>>(ps, hdri, hdrf);
  k_prep<<<NPTS / 256, 256, 0, stream>>>(gp, gpc, hdrf);
  k_vox<<<NPTS / 256, 256, 0, stream>>>(gpc, hdrf, hdri, vc);
  k_grid<<<1, 1, 0, stream>>>(hdri);
  k_hash<<<NPTS / 256, 256, 0, stream>>>(vc, hdri, hpt, tkey, tval, vhash);
  k_inv<<<NPTS / 256, 256, 0, stream>>>(hpt, tkey, tval, inv);
  k_enc<<<NPTS, FEAT, 0, stream>>>(gpc, w_in, b_in, ln_g, ln_b, inv, acc, vcnt);
  k_vfnorm<<<(NPTS * FEAT) / 256, 256, 0, stream>>>(acc, vcnt, hdri, vf32, vfb);
  k_nbr<<<NPTS / 256, 256, 0, stream>>>(vhash, tkey, tval, hdri, nbr);
  k_wpack<<<(NLAY * NTAP * FEAT * FEAT) / 256, 256, 0, stream>>>(cw1, cw2, wb1, wb2);

  const size_t wlsz = (size_t)NTAP * FEAT * FEAT;
  for (int l = 0; l < NLAY; l++) {
    k_conv<<<NPTS / 16, 256, 0, stream>>>(vfb, wb1 + l * wlsz, nbr, hdri, acc);
    hipMemsetAsync((char*)hdrf + HF_BNSUM * 4, 0, (size_t)(HF_BNSQ - HF_BNSUM + FEAT) * 4, stream);
    k_bnstats<<<NPTS / 64, 128, 0, stream>>>(acc, hdri, hdrf);
    k_bnapply<<<(NPTS * FEAT) / 256, 256, 0, stream>>>(acc, bn1g + l * FEAT, bn1b + l * FEAT,
                                                       hdri, hdrf, (const float*)nullptr,
                                                       (float*)nullptr, hb16);
    k_conv<<<NPTS / 16, 256, 0, stream>>>(hb16, wb2 + l * wlsz, nbr, hdri, acc);
    hipMemsetAsync((char*)hdrf + HF_BNSUM * 4, 0, (size_t)(HF_BNSQ - HF_BNSUM + FEAT) * 4, stream);
    k_bnstats<<<NPTS / 64, 128, 0, stream>>>(acc, hdri, hdrf);
    k_bnapply<<<(NPTS * FEAT) / 256, 256, 0, stream>>>(acc, bn2g + l * FEAT, bn2b + l * FEAT,
                                                       hdri, hdrf, vf32, vf32, vfb);
  }
  k_head<<<NPTS, NHID, 0, stream>>>(vf32, inv, wh1, bh1, wh2, bh2, out);
}
